// RBLCausalSelfAttention_31550829756966
// MI455X (gfx1250) — matrix operand order repaired and checked by reading, not compiled, not run on hardware
//
#include <hip/hip_runtime.h>

typedef unsigned short u16;
typedef unsigned int   u32;
typedef __attribute__((ext_vector_type(16))) __bf16 bf16x16;
typedef __attribute__((ext_vector_type(8)))  float  f32x8;

union Frag {
  bf16x16 v;
  uint4   q[2];
  u16     h[16];
};

__device__ inline u16 f2bf(float f) {
  u32 u = __float_as_uint(f);
  u32 r = u + 0x7FFFu + ((u >> 16) & 1u);   // round-to-nearest-even
  return (u16)(r >> 16);
}

__device__ inline f32x8 zero8() {
  f32x8 z;
#pragma unroll
  for (int i = 0; i < 8; ++i) z[i] = 0.0f;
  return z;
}

__device__ inline f32x8 wmma_bf16(bf16x16 a, bf16x16 b, f32x8 c) {
  // D = A(16x32) * B(32x16) + C(16x16), fp32 accum
  return __builtin_amdgcn_wmma_f32_16x16x32_bf16(false, a, false, b, (short)0, c,
                                                 false, false);
}

// async memory->LDS copy of 16 bytes per lane (CDNA5 GLOBAL_LOAD_ASYNC_TO_LDS_B128)
__device__ inline void async_copy_b128(u32 lds_addr, const void* gaddr) {
  asm volatile("global_load_async_to_lds_b128 %0, %1, off"
               :: "v"(lds_addr), "v"(gaddr) : "memory");
}
__device__ inline void async_wait0() {
  asm volatile("s_wait_asynccnt 0" ::: "memory");
}
__device__ inline u32 lds_addr_of(const void* p) {
  // flat LDS aperture keeps the LDS byte offset in addr[31:0]
  return (u32)(uintptr_t)p;
}

// ---------------------------------------------------------------------------
// fp32 -> bf16 conversion
// ---------------------------------------------------------------------------
__global__ __launch_bounds__(256) void f32_to_bf16_kernel(const float* __restrict__ in,
                                                          u16* __restrict__ out, int n) {
  int i = blockIdx.x * 256 + threadIdx.x;
  if (i < n) out[i] = f2bf(in[i]);
}

// ---------------------------------------------------------------------------
// C(MxN, fp32) = A(MxK, bf16 row-major) * B(KxN, bf16 row-major)
// Block tile 128x128, BK=32, double-buffered LDS.
// A tile staged with GLOBAL_LOAD_ASYNC_TO_LDS_B128 (ASYNCcnt pipeline),
// B tile staged through VGPRs (needs transpose on the way into LDS).
// 256 threads = 8 waves in a 2(M) x 4(N) grid; each wave = 4x2 WMMA tiles.
// ---------------------------------------------------------------------------
template <int N, int K>
__global__ __launch_bounds__(256) void gemm_bf16(const u16* __restrict__ A,
                                                 const u16* __restrict__ B,
                                                 float* __restrict__ C) {
  __shared__ __align__(16) u16 As[2][128][40];  // row-major M x K (pad 40)
  __shared__ __align__(16) u16 Bt[2][128][40];  // transposed: [n][k]

  const int t    = threadIdx.x;
  const int wave = t >> 5, lane = t & 31;
  const int wm = wave & 1, wn = wave >> 1;
  const int hi = lane >> 4, ln = lane & 15;
  const int m0 = blockIdx.y * 128, n0 = blockIdx.x * 128;

  // per-thread staging coordinates
  const int arow = t >> 1, ac16 = (t & 1) << 4;       // A: half-row per thread

  f32x8 acc[4][2];
#pragma unroll
  for (int mt = 0; mt < 4; ++mt)
#pragma unroll
    for (int nt = 0; nt < 2; ++nt) acc[mt][nt] = zero8();

  auto issueA = [&](int k0, int buf) {
    const u16* g = A + (size_t)(m0 + arow) * K + k0 + ac16;
    u32 l = lds_addr_of(&As[buf][arow][ac16]);
    async_copy_b128(l, g);
    async_copy_b128(l + 16, g + 8);
  };

  uint4 breg[2];
  auto loadB = [&](int k0) {
#pragma unroll
    for (int i = 0; i < 2; ++i) {
      int u  = t + i * 256;                 // 0..511
      int kk = u & 31;
      int n8 = (u >> 5) << 3;               // 0,8,...,120
      breg[i] = *(const uint4*)(B + (size_t)(k0 + kk) * N + n0 + n8);
    }
  };
  auto scatterB = [&](int buf) {
#pragma unroll
    for (int i = 0; i < 2; ++i) {
      int u  = t + i * 256;
      int kk = u & 31;
      int n8 = (u >> 5) << 3;
      const u16* e = (const u16*)&breg[i];
#pragma unroll
      for (int j = 0; j < 8; ++j) Bt[buf][n8 + j][kk] = e[j];
    }
  };

  // prologue: stage tile 0
  issueA(0, 0);
  loadB(0);
  scatterB(0);

  int buf = 0;
  for (int k0 = 0; k0 < K; k0 += 32) {
    async_wait0();          // this wave's async A stores into LDS are done
    __syncthreads();        // everyone's stores (A async + B ds) visible

    const bool next = (k0 + 32) < K;
    if (next) {             // overlap next tile's traffic with this tile's WMMAs
      issueA(k0 + 32, buf ^ 1);
      loadB(k0 + 32);
    }

    // --- A fragments: lane ln = row, halves hold K 0-15 / 16-31
    Frag a[4];
#pragma unroll
    for (int mt = 0; mt < 4; ++mt) {
      int row  = wm * 64 + mt * 16 + ln;
      int koff = hi * 16;
      a[mt].q[0] = *(const uint4*)&As[buf][row][koff];
      a[mt].q[1] = *(const uint4*)&As[buf][row][koff + 8];
    }
    // --- B fragments: lane ln = column n, halves hold K 0-15 / 16-31
    Frag b[2];
#pragma unroll
    for (int nt = 0; nt < 2; ++nt) {
      int n     = wn * 32 + nt * 16 + ln;
      int koff2 = hi * 16;
      b[nt].q[0] = *(const uint4*)&Bt[buf][n][koff2];
      b[nt].q[1] = *(const uint4*)&Bt[buf][n][koff2 + 8];
    }
#pragma unroll
    for (int mt = 0; mt < 4; ++mt)
#pragma unroll
      for (int nt = 0; nt < 2; ++nt)
        acc[mt][nt] = wmma_bf16(a[mt].v, b[nt].v, acc[mt][nt]);

    if (next) scatterB(buf ^ 1);   // into the other buffer; guarded by next barrier
    buf ^= 1;
  }

  // --- store C (C/D layout: VGPR j -> rows j / j+8 across lane halves)
  // N is a template constant: row strides fold into immediate store offsets.
#pragma unroll
  for (int mt = 0; mt < 4; ++mt) {
    const int row = m0 + wm * 64 + mt * 16 + hi * 8;
#pragma unroll
    for (int nt = 0; nt < 2; ++nt) {
      float* cp = C + (size_t)row * N + n0 + wn * 32 + nt * 16 + ln;
#pragma unroll
      for (int j = 0; j < 8; ++j) cp[(size_t)j * N] = acc[mt][nt][j];
    }
  }
}

// ---------------------------------------------------------------------------
// qkv post-process: unit-norm q,k; RoPE on even heads; emit bf16 (B,H,S,D)
// One block per token (b,s); 16 threads per head.
// ---------------------------------------------------------------------------
__global__ __launch_bounds__(256) void qkv_prep(const float* __restrict__ qkv,
                                                u16* __restrict__ qo,
                                                u16* __restrict__ ko,
                                                u16* __restrict__ vo) {
  const int blk = blockIdx.x;            // b*2048 + s
  const int tok = blk & 2047;
  const int b   = blk >> 11;
  const int t   = threadIdx.x;
  const int h   = t >> 4, l16 = t & 15;

  const float* base = qkv + (size_t)blk * 6144 + h * 128;

  float qv[8], kv[8], vv[8];
  float sq = 0.0f, sk = 0.0f;
#pragma unroll
  for (int i = 0; i < 4; ++i) {
    int p = l16 + 16 * i;                // pair index 0..63
    float q1 = base[p],        q2 = base[p + 64];
    float k1 = base[2048 + p], k2 = base[2048 + p + 64];
    qv[i] = q1; qv[i + 4] = q2;
    kv[i] = k1; kv[i + 4] = k2;
    vv[i] = base[4096 + p]; vv[i + 4] = base[4096 + p + 64];
    sq += q1 * q1 + q2 * q2;
    sk += k1 * k1 + k2 * k2;
  }
  // reduce over the 16 lanes of this head (stays inside a 16-lane half)
  for (int m = 8; m >= 1; m >>= 1) {
    sq += __shfl_xor(sq, m, 32);
    sk += __shfl_xor(sk, m, 32);
  }
  const float rq = rsqrtf(sq + 1e-6f);
  const float rk = rsqrtf(sk + 1e-6f);
  const bool  dorope = (h & 1) == 0;     // ROPE_MASK = [1,0,1,0,...]

  const size_t obase = (((size_t)b * 16 + h) * 2048 + tok) * 128;
#pragma unroll
  for (int i = 0; i < 4; ++i) {
    int p = l16 + 16 * i;
    float q1 = qv[i] * rq, q2 = qv[i + 4] * rq;
    float k1 = kv[i] * rk, k2 = kv[i + 4] * rk;
    if (dorope) {
      // inv_freq = 10000^(-p/64); theta = tok * inv_freq
      float inv = expf(-(float)p * (9.210340371976184f / 64.0f));
      float th  = (float)tok * inv;
      float c = cosf(th), sn = sinf(th);
      float a1 = q1 * c - q2 * sn, a2 = q1 * sn + q2 * c;
      q1 = a1; q2 = a2;
      float b1 = k1 * c - k2 * sn, b2 = k1 * sn + k2 * c;
      k1 = b1; k2 = b2;
    }
    qo[obase + p]      = f2bf(q1);
    qo[obase + p + 64] = f2bf(q2);
    ko[obase + p]      = f2bf(k1);
    ko[obase + p + 64] = f2bf(k2);
    vo[obase + p]      = f2bf(vv[i]);
    vo[obase + p + 64] = f2bf(vv[i + 4]);
  }
}

// ---------------------------------------------------------------------------
// Flash attention: block = (b, h, 128-query-row tile). 8 waves x 16 rows.
// q fragments in registers; K tile async-staged into LDS; V staged through
// VGPRs (transposed); online softmax; P re-laid out via per-wave LDS staging.
// Output y bf16 in (B, S, H*D) layout for the final GEMM.
// ---------------------------------------------------------------------------
__global__ __launch_bounds__(256) void attn_kernel(const u16* __restrict__ qb_,
                                                   const u16* __restrict__ kb_,
                                                   const u16* __restrict__ vb_,
                                                   const float* __restrict__ gains,
                                                   u16* __restrict__ yb) {
  __shared__ __align__(16) u16 ks[64][136];     // [key][d]     (17.4 KB)
  __shared__ __align__(16) u16 vt[128][72];     // [d][key]     (18.4 KB)
  __shared__ __align__(16) u16 ps[8][16][72];   // per-wave P   (18.4 KB)

  const int blk = blockIdx.x;
  const int qb = blk & 15;                 // S/128 = 16 query tiles
  const int h  = (blk >> 4) & 15;
  const int b  = blk >> 8;

  const int t    = threadIdx.x;
  const int wave = t >> 5, lane = t & 31;
  const int hi = lane >> 4, ln = lane & 15;

  const size_t bh   = ((size_t)b * 16 + h) * 2048;
  const u16*  qptr  = qb_ + bh * 128;
  const float gain  = gains[h];
  const int   row0  = qb * 128 + wave * 16;    // this wave's first query row

  // preload this wave's q A-fragments (16 rows x 128 d = 4 k-steps)
  Frag qf[4];
  {
    const u16* rp = qptr + (size_t)(row0 + ln) * 128;
    int koff = hi * 16;
#pragma unroll
    for (int kk = 0; kk < 4; ++kk) {
      qf[kk].q[0] = *(const uint4*)(rp + kk * 32 + koff);
      qf[kk].q[1] = *(const uint4*)(rp + kk * 32 + koff + 8);
    }
  }

  f32x8 o[8];
#pragma unroll
  for (int dt = 0; dt < 8; ++dt) o[dt] = zero8();
  float mrun[8], lrun[8];
#pragma unroll
  for (int e = 0; e < 8; ++e) { mrun[e] = -1e30f; lrun[e] = 0.0f; }

  const int jmax = 2 * qb + 1;             // causal clip: keys <= qb*128+127
  for (int j = 0; j <= jmax; ++j) {
    __syncthreads();                       // previous iteration's readers done
    // --- K rows: async memory->LDS (row-major, no transpose needed)
#pragma unroll
    for (int i = 0; i < 4; ++i) {
      int u   = t + i * 256;               // 0..1023
      int key = u >> 4;
      int d8  = (u & 15) << 3;
      const u16* g = kb_ + (bh + (size_t)j * 64 + key) * 128 + d8;
      async_copy_b128(lds_addr_of(&ks[key][d8]), g);
    }
    // --- V rows: through VGPRs, transposed scatter into [d][key]
#pragma unroll
    for (int i = 0; i < 4; ++i) {
      int u   = t + i * 256;
      int key = u >> 4;
      int d8  = (u & 15) << 3;
      uint4 vv = *(const uint4*)(vb_ + (bh + (size_t)j * 64 + key) * 128 + d8);
      const u16* e = (const u16*)&vv;
#pragma unroll
      for (int q2 = 0; q2 < 8; ++q2) vt[d8 + q2][key] = e[q2];
    }
    async_wait0();
    __syncthreads();

    // logits: 16 rows x 64 keys, contraction over d (4 WMMA k-steps)
    f32x8 s[4];
#pragma unroll
    for (int nt = 0; nt < 4; ++nt) s[nt] = zero8();
#pragma unroll
    for (int kk = 0; kk < 4; ++kk) {
#pragma unroll
      for (int nt = 0; nt < 4; ++nt) {
        Frag bf;
        int n     = nt * 16 + ln;
        int koff2 = hi * 16;
        bf.q[0] = *(const uint4*)&ks[n][kk * 32 + koff2];
        bf.q[1] = *(const uint4*)&ks[n][kk * 32 + koff2 + 8];
        s[nt] = wmma_bf16(qf[kk].v, bf.v, s[nt]);
      }
    }

    // gain, causal mask, row max (rows live in 16-lane halves)
    float rmax[8];
#pragma unroll
    for (int e = 0; e < 8; ++e) {
      int rowg = row0 + e + hi * 8;
      float mx = -1e30f;
#pragma unroll
      for (int nt = 0; nt < 4; ++nt) {
        int key = j * 64 + nt * 16 + ln;
        float v = s[nt][e] * gain;
        if (key > rowg) v = -1e30f;
        s[nt][e] = v;
        mx = fmaxf(mx, v);
      }
      for (int m = 8; m >= 1; m >>= 1) mx = fmaxf(mx, __shfl_xor(mx, m, 32));
      rmax[e] = mx;
    }

    // online softmax update + P in bf16 staged to per-wave LDS
#pragma unroll
    for (int e = 0; e < 8; ++e) {
      float mnew  = fmaxf(mrun[e], rmax[e]);
      float scale = __expf(mrun[e] - mnew);
      float psum  = 0.0f;
#pragma unroll
      for (int nt = 0; nt < 4; ++nt) {
        float p = __expf(s[nt][e] - mnew);
        s[nt][e] = p;
        psum += p;
      }
      for (int m = 8; m >= 1; m >>= 1) psum += __shfl_xor(psum, m, 32);
      lrun[e] = lrun[e] * scale + psum;
      mrun[e] = mnew;
#pragma unroll
      for (int dt = 0; dt < 8; ++dt) o[dt][e] *= scale;
#pragma unroll
      for (int nt = 0; nt < 4; ++nt)
        ps[wave][e + hi * 8][nt * 16 + ln] = f2bf(s[nt][e]);
    }
    // same-wave LDS ops are in-order: safe to read ps back now.

    // O += P(16x64) * V(64x128): 2 key-steps x 8 d-tiles
#pragma unroll
    for (int st = 0; st < 2; ++st) {
      Frag pa;
      int koff = hi * 16;
      pa.q[0] = *(const uint4*)&ps[wave][ln][st * 32 + koff];
      pa.q[1] = *(const uint4*)&ps[wave][ln][st * 32 + koff + 8];
#pragma unroll
      for (int dt = 0; dt < 8; ++dt) {
        Frag vb;
        int d     = dt * 16 + ln;
        int koff2 = hi * 16;
        vb.q[0] = *(const uint4*)&vt[d][st * 32 + koff2];
        vb.q[1] = *(const uint4*)&vt[d][st * 32 + koff2 + 8];
        o[dt] = wmma_bf16(pa.v, vb.v, o[dt]);
      }
    }
  }

  // epilogue: normalize and emit bf16 y[b][row][h*128 + d]
#pragma unroll
  for (int dt = 0; dt < 8; ++dt)
#pragma unroll
    for (int e = 0; e < 8; ++e) {
      int rowg = row0 + e + hi * 8;
      float val = o[dt][e] / lrun[e];
      yb[((size_t)b * 2048 + rowg) * 2048 + h * 128 + dt * 16 + ln] = f2bf(val);
    }
}

// ---------------------------------------------------------------------------
// Host-side orchestration
// ---------------------------------------------------------------------------
extern "C" void kernel_launch(void* const* d_in, const int* in_sizes, int n_in,
                              void* d_out, int out_size, void* d_ws, size_t ws_size,
                              hipStream_t stream) {
  const float* x      = (const float*)d_in[0];  // (2,2048,2048)
  const float* w_qkv  = (const float*)d_in[1];  // (2048,6144)
  const float* w_out  = (const float*)d_in[2];  // (2048,2048)
  const float* q_gain = (const float*)d_in[3];  // (16,)
  float* out = (float*)d_out;                   // (2,2048,2048)

  const int NX = 2 * 2048 * 2048;   // 8388608
  const int NW = 2048 * 6144;       // 12582912
  const int NO = 2048 * 2048;       // 4194304
  const size_t MB = 1024ull * 1024ull;

  char* ws = (char*)d_ws;
  u16*   xb    = (u16*)(ws);               //  16 MB
  u16*   wqkvb = (u16*)(ws + 16 * MB);     //  24 MB
  u16*   woutb = (u16*)(ws + 40 * MB);     //   8 MB
  float* qkv   = (float*)(ws + 48 * MB);   //  96 MB
  u16*   qbuf  = (u16*)(ws + 144 * MB);    //  16 MB
  u16*   kbuf  = (u16*)(ws + 160 * MB);    //  16 MB
  u16*   vbuf  = (u16*)(ws + 176 * MB);    //  16 MB
  u16*   ybuf  = (u16*)(ws + 192 * MB);    //  16 MB -> 208 MB total

  f32_to_bf16_kernel<<<(NX + 255) / 256, 256, 0, stream>>>(x, xb, NX);
  f32_to_bf16_kernel<<<(NW + 255) / 256, 256, 0, stream>>>(w_qkv, wqkvb, NW);
  f32_to_bf16_kernel<<<(NO + 255) / 256, 256, 0, stream>>>(w_out, woutb, NO);

  // qkv = x @ w_qkv : M=4096, N=6144, K=2048
  gemm_bf16<6144, 2048><<<dim3(48, 32), 256, 0, stream>>>(xb, wqkvb, qkv);
  // norm + rope + split into (B,H,S,D) bf16
  qkv_prep<<<2 * 2048, 256, 0, stream>>>(qkv, qbuf, kbuf, vbuf);
  // attention: B*H*(S/128) = 512 blocks
  attn_kernel<<<512, 256, 0, stream>>>(qbuf, kbuf, vbuf, q_gain, ybuf);
  // out = y @ w_out : M=4096, N=2048, K=2048
  gemm_bf16<2048, 2048><<<dim3(16, 32), 256, 0, stream>>>(ybuf, woutb, out);
}
